// ProbAttention_42923903156803
// MI455X (gfx1250) — matrix operand order repaired and checked by reading, not compiled, not run on hardware
//
#include <hip/hip_runtime.h>

// ---------------------------------------------------------------------------
// ProbSparse attention (Informer) for MI455X / gfx1250.
// Shapes: B=4, T=4, L=2048, D=512, H=8, KD=VD=64, U=40.
// Projections use v_wmma_f32_16x16x32_bf16 (fp32->bf16 inputs, f32 accum).
// ---------------------------------------------------------------------------

#define BB 4
#define TT 4
#define LL 2048
#define DD 512
#define HH 8
#define EE 64      // KD == VD
#define UU 40
#define BT (BB*TT)          // 16
#define NCOL (HH*EE)        // 512 (== DD here)
#define NCHUNK 16           // scan chunks over L
#define CLEN (LL/NCHUNK)    // 128 rows per chunk

typedef __attribute__((ext_vector_type(16))) __bf16 v16bf;
typedef __attribute__((ext_vector_type(8)))  float  v8f;

union AFrag { v16bf v; unsigned int u[8]; };

static __device__ __forceinline__ unsigned short f2bf(float f) {
    unsigned int u = __float_as_uint(f);
    u += 0x7FFFu + ((u >> 16) & 1u);       // round-to-nearest-even
    return (unsigned short)(u >> 16);
}

// ---------------------------------------------------------------------------
// Kernel 1: projection GEMM.  C[2048x512] = X[2048x512] * W[512x512]
// grid = (L/64, NCOL/128, BT*2), block = 256 (8 waves).
// z = bt*2 + qv ; qv==0 -> Q = queries*Wq ; qv==1 -> V = values*Wv
// ---------------------------------------------------------------------------
__global__ __launch_bounds__(256)
void proj_gemm_kernel(const float* __restrict__ queries,
                      const float* __restrict__ values,
                      const float* __restrict__ Wq,
                      const float* __restrict__ Wv,
                      float* __restrict__ Qw,
                      float* __restrict__ Vw)
{
    __shared__ unsigned short As[64][34];    // [m][k] bf16, pad stride 34
    __shared__ unsigned short Bs[128][34];   // [n][k] bf16 (W tile transposed)

    const int tid  = threadIdx.x;
    const int lane = tid & 31;
    const int wave = tid >> 5;

    const int m0 = blockIdx.x * 64;
    const int n0 = blockIdx.y * 128;
    const int z  = blockIdx.z;
    const int bt = z >> 1;
    const int qv = z & 1;

    const float* __restrict__ X = (qv == 0 ? queries : values) + (size_t)bt * LL * DD;
    const float* __restrict__ W = (qv == 0 ? Wq : Wv);
    float*       __restrict__ O = (qv == 0 ? Qw : Vw) + (size_t)bt * LL * NCOL;

    // wave -> (16-row strip, 64-col strip)
    const int wm = (wave >> 1) * 16;     // 0,16,32,48
    const int wn = (wave & 1) * 64;      // 0,64

    v8f acc[4];
    #pragma unroll
    for (int t = 0; t < 4; ++t) acc[t] = (v8f){0,0,0,0,0,0,0,0};

    const int kh = lane >> 4;            // K-half select (wave32 layout)
    const int ml = lane & 15;

    for (int k0 = 0; k0 < DD; k0 += 32) {
        // ---- stage A tile 64x32 (fp32 -> bf16) ----
        {
            const int ka = tid & 31;
            const int ra = tid >> 5;            // 8 rows / pass
            #pragma unroll
            for (int p = 0; p < 8; ++p) {
                const int m = ra + p * 8;
                As[m][ka] = f2bf(X[(size_t)(m0 + m) * DD + k0 + ka]);
            }
        }
        // ---- stage B tile 32x128, transposed into [n][k] ----
        {
            const int nb = tid & 127;
            const int kb = tid >> 7;            // 2 k-rows / pass
            #pragma unroll
            for (int p = 0; p < 16; ++p) {
                const int k = kb + p * 2;
                Bs[nb][k] = f2bf(W[(size_t)(k0 + k) * NCOL + n0 + nb]);
            }
        }
        __syncthreads();

        // ---- A fragment: 16x32 bf16, documented wave32 layout ----
        AFrag a;
        #pragma unroll
        for (int v = 0; v < 8; ++v) {
            const int kk = ((v >> 2) << 4) + kh * 8 + ((v & 3) << 1);
            a.u[v] = *(const unsigned int*)&As[wm + ml][kk];
        }
        // ---- 4 B fragments (32x16 each) + WMMA ----
        #pragma unroll
        for (int t = 0; t < 4; ++t) {
            AFrag b;
            const int n = wn + t * 16 + ml;
            #pragma unroll
            for (int v = 0; v < 8; ++v) {
                const int kk = ((v >> 2) << 4) + kh * 8 + ((v & 3) << 1);
                b.u[v] = *(const unsigned int*)&Bs[n][kk];
            }
            acc[t] = __builtin_amdgcn_wmma_f32_16x16x32_bf16(
                false, a.v, false, b.v, (short)0, acc[t], false, false);
        }
        __syncthreads();
    }

    // ---- store C: VGPR r -> row (lane>=16 ? 8 : 0)+r, col = lane&15 ----
    const int rbase = m0 + wm + kh * 8;
    #pragma unroll
    for (int t = 0; t < 4; ++t) {
        const int col = n0 + wn + t * 16 + ml;
        #pragma unroll
        for (int r = 0; r < 8; ++r) {
            O[(size_t)(rbase + r) * NCOL + col] = acc[t][r];
        }
    }
}

// ---------------------------------------------------------------------------
// Kernel 2: sampled score metric  M[l] = max_k QKs - sum_k QKs / L
// grid = (L/256, BT*H), block = 256; one l per thread.
// ---------------------------------------------------------------------------
__global__ __launch_bounds__(256)
void sample_score_kernel(const float* __restrict__ Qw,
                         const float* __restrict__ Vw,
                         const int*   __restrict__ idxs,
                         float*       __restrict__ Mw)
{
    __shared__ float Ks[UU * EE];   // 40 sampled V rows for this head

    const int tid = threadIdx.x;
    const int bth = blockIdx.y;
    const int bt  = bth >> 3;
    const int h   = bth & 7;
    const size_t vbase = (size_t)bt * LL * NCOL;

    for (int i = tid; i < UU * EE; i += 256) {
        const int k = i >> 6;
        const int j = i & 63;
        Ks[i] = Vw[vbase + (size_t)idxs[k] * NCOL + h * EE + j];
    }
    __syncthreads();

    const int l = blockIdx.x * 256 + tid;
    const float4* q4 = (const float4*)(Qw + vbase + (size_t)l * NCOL + h * EE);

    float4 q[16];
    #pragma unroll
    for (int j = 0; j < 16; ++j) q[j] = q4[j];

    float mx = -3.0e38f, sm = 0.0f;
    for (int k = 0; k < UU; ++k) {
        const float4* ks = (const float4*)&Ks[k * EE];
        float d = 0.0f;
        #pragma unroll
        for (int j = 0; j < 16; ++j) {
            float4 a = q[j], b = ks[j];
            d += a.x * b.x + a.y * b.y + a.z * b.z + a.w * b.w;
        }
        mx = fmaxf(mx, d);
        sm += d;
    }
    Mw[(size_t)bth * LL + l] = mx - sm * (1.0f / (float)LL);
}

// ---------------------------------------------------------------------------
// Kernel 3: per-(b,t,h) top-U selection over L (iterative argmax).
// grid = BT*H, block = 256.
// ---------------------------------------------------------------------------
__global__ __launch_bounds__(256)
void topk_kernel(const float* __restrict__ Mw, int* __restrict__ Ti)
{
    __shared__ float mv[LL];
    __shared__ float rv[256];
    __shared__ int   ri[256];

    const int tid = threadIdx.x;
    const int bth = blockIdx.x;

    for (int i = tid; i < LL; i += 256) mv[i] = Mw[(size_t)bth * LL + i];
    __syncthreads();

    for (int it = 0; it < UU; ++it) {
        float bv = -3.0e38f; int bi = LL;
        for (int i = tid; i < LL; i += 256) {
            const float v = mv[i];
            if (v > bv || (v == bv && i < bi)) { bv = v; bi = i; }
        }
        rv[tid] = bv; ri[tid] = bi;
        __syncthreads();
        for (int off = 128; off > 0; off >>= 1) {
            if (tid < off) {
                const float v2 = rv[tid + off]; const int i2 = ri[tid + off];
                if (v2 > rv[tid] || (v2 == rv[tid] && i2 < ri[tid])) {
                    rv[tid] = v2; ri[tid] = i2;
                }
            }
            __syncthreads();
        }
        if (tid == 0) { Ti[bth * UU + it] = ri[0]; mv[ri[0]] = -3.0e38f; }
        __syncthreads();
    }
}

// ---------------------------------------------------------------------------
// Kernel 4a: scores S[bth][u][s] = (Qred[u].V[s] * tau + delta) / 8
// One V pass for all 40 u.  grid = (L/256, BT*H), block = 256.
// ---------------------------------------------------------------------------
__global__ __launch_bounds__(256)
void scores_kernel(const float* __restrict__ Qw,
                   const float* __restrict__ Vw,
                   const int*   __restrict__ Ti,
                   const float* __restrict__ tau,
                   const float* __restrict__ delta,
                   float*       __restrict__ Sw)
{
    __shared__ float qr[UU * EE];   // 40 selected Q rows for this head

    const int tid = threadIdx.x;
    const int bth = blockIdx.y;
    const int bt  = bth >> 3;
    const int h   = bth & 7;
    const size_t base = (size_t)bt * LL * NCOL;

    for (int i = tid; i < UU * EE; i += 256) {
        const int u = i >> 6;
        const int j = i & 63;
        qr[i] = Qw[base + (size_t)Ti[bth * UU + u] * NCOL + h * EE + j];
    }
    __syncthreads();

    const float tauv   = tau[0];
    const float deltav = delta[0];

    const int s = blockIdx.x * 256 + tid;
    const float4* v4 = (const float4*)(Vw + base + (size_t)s * NCOL + h * EE);
    float4 v[16];
    #pragma unroll
    for (int j = 0; j < 16; ++j) v[j] = v4[j];

    float* srow = Sw + (size_t)bth * UU * LL;
    for (int u = 0; u < UU; ++u) {
        const float4* qk = (const float4*)&qr[u * EE];
        float d = 0.0f;
        #pragma unroll
        for (int j = 0; j < 16; ++j) {
            float4 a = qk[j], b = v[j];
            d += a.x * b.x + a.y * b.y + a.z * b.z + a.w * b.w;
        }
        srow[(size_t)u * LL + s] = (d * tauv + deltav) * 0.125f;
    }
}

// ---------------------------------------------------------------------------
// Kernel 4b: row softmax (normalized in place).  grid = BT*H*U, block = 256.
// ---------------------------------------------------------------------------
__global__ __launch_bounds__(256)
void softmax_kernel(float* __restrict__ Sw)
{
    __shared__ float red[256];

    const int tid = threadIdx.x;
    float* row = Sw + (size_t)blockIdx.x * LL;

    float x[8];
    float lmax = -3.0e38f;
    #pragma unroll
    for (int i = 0; i < 8; ++i) {
        x[i] = row[tid + i * 256];
        lmax = fmaxf(lmax, x[i]);
    }
    red[tid] = lmax; __syncthreads();
    for (int off = 128; off > 0; off >>= 1) {
        if (tid < off) red[tid] = fmaxf(red[tid], red[tid + off]);
        __syncthreads();
    }
    const float mx = red[0];
    __syncthreads();

    float lsum = 0.0f;
    #pragma unroll
    for (int i = 0; i < 8; ++i) {
        x[i] = __expf(x[i] - mx);
        lsum += x[i];
    }
    red[tid] = lsum; __syncthreads();
    for (int off = 128; off > 0; off >>= 1) {
        if (tid < off) red[tid] += red[tid + off];
        __syncthreads();
    }
    const float inv = 1.0f / red[0];

    #pragma unroll
    for (int i = 0; i < 8; ++i) row[tid + i * 256] = x[i] * inv;
}

// ---------------------------------------------------------------------------
// Kernel 4c: attn[bth][u][e] = sum_s P[u][s] * V[s][e]  (one V pass, 40 accums)
// grid = BT*H, block = 256 (c = tid>>6 chunks of s, e = tid&63).
// ---------------------------------------------------------------------------
__global__ __launch_bounds__(256)
void attn_kernel(const float* __restrict__ Vw,
                 const float* __restrict__ Sw,
                 float*       __restrict__ Aw)
{
    __shared__ float part[4][EE];

    const int tid = threadIdx.x;
    const int bth = blockIdx.x;
    const int bt  = bth >> 3;
    const int h   = bth & 7;
    const size_t base = (size_t)bt * LL * NCOL;
    const float* P = Sw + (size_t)bth * UU * LL;

    const int e = tid & 63;
    const int c = tid >> 6;

    float acc[UU];
    #pragma unroll
    for (int u = 0; u < UU; ++u) acc[u] = 0.0f;

    for (int j = 0; j < LL / 4; ++j) {
        const int s = c * (LL / 4) + j;
        const float v = Vw[base + (size_t)s * NCOL + h * EE + e];
        #pragma unroll
        for (int u = 0; u < UU; ++u) acc[u] += P[(size_t)u * LL + s] * v;
    }

    for (int u = 0; u < UU; ++u) {
        part[c][e] = acc[u];
        __syncthreads();
        if (tid < EE) {
            Aw[((size_t)bth * UU + u) * EE + tid] =
                part[0][tid] + part[1][tid] + part[2][tid] + part[3][tid];
        }
        __syncthreads();
    }
}

// ---------------------------------------------------------------------------
// Kernel 5pre: replacement map repl[bth][l] = u (or -1).  grid = BT*H.
// ---------------------------------------------------------------------------
__global__ __launch_bounds__(256)
void repl_kernel(const int* __restrict__ Ti, int* __restrict__ repl)
{
    const int tid = threadIdx.x;
    const int bth = blockIdx.x;
    for (int i = tid; i < LL; i += 256) repl[(size_t)bth * LL + i] = -1;
    __syncthreads();
    if (tid < UU) repl[(size_t)bth * LL + Ti[bth * UU + tid]] = tid;
}

// ---------------------------------------------------------------------------
// Kernel 5a: chunk sums csum[bth][c][e] = sum_{l in chunk c} V[l][h][e]
// grid = (NCHUNK, BT*H), block = 64.
// ---------------------------------------------------------------------------
__global__ __launch_bounds__(64)
void chunksum_kernel(const float* __restrict__ Vw, float* __restrict__ csum)
{
    const int e   = threadIdx.x;
    const int c   = blockIdx.x;
    const int bth = blockIdx.y;
    const int bt  = bth >> 3;
    const int h   = bth & 7;
    const size_t base = (size_t)bt * LL * NCOL + h * EE + e;

    float s = 0.0f;
    for (int j = 0; j < CLEN; ++j)
        s += Vw[base + (size_t)(c * CLEN + j) * NCOL];
    csum[((size_t)bth * NCHUNK + c) * EE + e] = s;
}

// ---------------------------------------------------------------------------
// Kernel 5b: per-chunk inclusive scan with row replacement; writes per-head
// weighted contribution ctxw[bth][l][e] = w_out[h] * ctx[l][e].
// grid = (NCHUNK, BT*H), block = 64.
// ---------------------------------------------------------------------------
__global__ __launch_bounds__(64)
void scan_kernel(const float* __restrict__ Vw,
                 const float* __restrict__ csum,
                 const int*   __restrict__ repl,
                 const float* __restrict__ Aw,
                 const float* __restrict__ w_out,
                 float*       __restrict__ ctxw)
{
    const int e   = threadIdx.x;
    const int c   = blockIdx.x;
    const int bth = blockIdx.y;
    const int bt  = bth >> 3;
    const int h   = bth & 7;
    const size_t vbase = (size_t)bt * LL * NCOL + h * EE + e;
    const float wv = w_out[h];

    float acc = 0.0f;                       // exclusive prefix of earlier chunks
    for (int cp = 0; cp < c; ++cp)
        acc += csum[((size_t)bth * NCHUNK + cp) * EE + e];

    for (int j = 0; j < CLEN; ++j) {
        const int l = c * CLEN + j;
        acc += Vw[vbase + (size_t)l * NCOL];
        const int u = repl[(size_t)bth * LL + l];
        const float val = (u >= 0) ? Aw[((size_t)bth * UU + u) * EE + e] : acc;
        ctxw[((size_t)bth * LL + l) * EE + e] = val * wv;
    }
}

// ---------------------------------------------------------------------------
// Kernel 5c: out[bt][l][e] = sum_h ctxw[bt*8+h][l][e].
// grid = BT*L*EE/256, block = 256.
// ---------------------------------------------------------------------------
__global__ __launch_bounds__(256)
void reduce_out_kernel(const float* __restrict__ ctxw, float* __restrict__ out)
{
    const size_t g = (size_t)blockIdx.x * 256 + threadIdx.x;   // over BT*L*EE
    const int e  = (int)(g & 63);
    const int l  = (int)((g >> 6) & (LL - 1));
    const int bt = (int)(g >> 17);

    float s = 0.0f;
    #pragma unroll
    for (int h = 0; h < HH; ++h)
        s += ctxw[(((size_t)(bt * HH + h)) * LL + l) * EE + e];
    out[g] = s;
}

// ---------------------------------------------------------------------------
// Launch
// ---------------------------------------------------------------------------
extern "C" void kernel_launch(void* const* d_in, const int* in_sizes, int n_in,
                              void* d_out, int out_size, void* d_ws, size_t ws_size,
                              hipStream_t stream)
{
    const float* queries = (const float*)d_in[0];
    // d_in[1] = keys   (unused: reference deletes K)
    const float* values  = (const float*)d_in[2];
    const float* Wq      = (const float*)d_in[3];
    // d_in[4] = Wk     (unused)
    const float* Wv      = (const float*)d_in[5];
    const float* w_out   = (const float*)d_in[6];
    const float* tau     = (const float*)d_in[7];
    const float* delta   = (const float*)d_in[8];
    const int*   idxs    = (const int*)  d_in[9];
    float*       out     = (float*)d_out;

    // ---- workspace layout (floats unless noted) ----
    // Qw, Vw:   BT*L*NCOL each (64 MB each)
    // SC:       shared scratch, max(S, ctxw) = BT*H*L*EE floats (64 MB)
    //           S    = scores/probs [bth][u][s]  (40 MB, used by 4a/4b/4c)
    //           ctxw = per-head context plane    (64 MB, used by 5b/5c)
    // Mw:       BT*H*L ; csum: BT*H*NCHUNK*EE ; Aw: BT*H*U*EE
    // Ti (int): BT*H*U ; repl (int): BT*H*L
    float* Qw   = (float*)d_ws;
    float* Vw   = Qw + (size_t)BT * LL * NCOL;
    float* SC   = Vw + (size_t)BT * LL * NCOL;
    float* Mw   = SC + (size_t)BT * HH * LL * EE;
    float* csum = Mw + (size_t)BT * HH * LL;
    float* Aw   = csum + (size_t)BT * HH * NCHUNK * EE;
    int*   Ti   = (int*)(Aw + (size_t)BT * HH * UU * EE);
    int*   repl = Ti + (size_t)BT * HH * UU;

    float* Sw   = SC;   // scores buffer (lifetime: 4a..4c)
    float* ctxw = SC;   // context plane (lifetime: 5b..5c)

    {   // 1. projections (bf16 WMMA)
        dim3 g(LL / 64, NCOL / 128, BT * 2);
        proj_gemm_kernel<<<g, 256, 0, stream>>>(queries, values, Wq, Wv, Qw, Vw);
    }
    {   // 2. sampled score metric
        dim3 g(LL / 256, BT * HH);
        sample_score_kernel<<<g, 256, 0, stream>>>(Qw, Vw, idxs, Mw);
    }
    {   // 3. top-U selection
        topk_kernel<<<dim3(BT * HH), 256, 0, stream>>>(Mw, Ti);
    }
    {   // 4a. all-u scores in one V pass
        dim3 g(LL / 256, BT * HH);
        scores_kernel<<<g, 256, 0, stream>>>(Qw, Vw, Ti, tau, delta, Sw);
    }
    {   // 4b. row softmax (normalized)
        softmax_kernel<<<dim3(BT * HH * UU), 256, 0, stream>>>(Sw);
    }
    {   // 4c. attn = P * V, one V pass
        attn_kernel<<<dim3(BT * HH), 256, 0, stream>>>(Vw, Sw, Aw);
    }
    {   // 5pre. replacement map
        repl_kernel<<<dim3(BT * HH), 256, 0, stream>>>(Ti, repl);
    }
    {   // 5a. chunk partial sums
        dim3 g(NCHUNK, BT * HH);
        chunksum_kernel<<<g, 64, 0, stream>>>(Vw, csum);
    }
    {   // 5b. parallel scan + replacement + head weight
        dim3 g(NCHUNK, BT * HH);
        scan_kernel<<<g, 64, 0, stream>>>(Vw, csum, repl, Aw, w_out, ctxw);
    }
    {   // 5c. reduce over heads -> out
        reduce_out_kernel<<<dim3((BT * LL * EE) / 256), 256, 0, stream>>>(ctxw, out);
    }
}
